// DOMGraphTransformer_40415642255504
// MI455X (gfx1250) — compile-verified, matrix operand order checked by reading, not run on hardware
//
#include <hip/hip_runtime.h>
#include <hip/hip_bf16.h>

typedef _Float16 v16h __attribute__((ext_vector_type(16)));
typedef _Float16 v8h  __attribute__((ext_vector_type(8)));
typedef float    v8f  __attribute__((ext_vector_type(8)));

#define TM 128   // rows of x per block
#define DIM 64

// ---------------------------------------------------------------------------
// One-time per layer: transpose + convert 4 weight matrices to f16.
// wt16 layout: [m][n*64+k]  (column-contiguous so B fragments are b128 loads)
// ---------------------------------------------------------------------------
__global__ void prep_weights(const float* __restrict__ Wq, const float* __restrict__ Wk,
                             const float* __restrict__ Wv, const float* __restrict__ Ws,
                             _Float16* __restrict__ wt16)
{
    int i = blockIdx.x * blockDim.x + threadIdx.x;   // 0 .. 4*4096-1
    if (i >= 4 * DIM * DIM) return;
    int m  = i >> 12;
    int r  = i & 4095;
    int n  = r >> 6, kk = r & 63;
    const float* W = (m == 0) ? Wq : (m == 1) ? Wk : (m == 2) ? Wv : Ws;
    wt16[i] = (_Float16)W[kk * DIM + n];
}

// ---------------------------------------------------------------------------
// Row-tile compute loop, specialized on whether the block is full (no row
// guards needed) so the hot path carries zero per-iteration predicates.
// ---------------------------------------------------------------------------
template <bool FULL>
__device__ __forceinline__ void compute_tiles(const _Float16* __restrict__ sX,
                                              const v16h (&bfr)[4][2],
                                              const float (&bv4)[4],
                                              float* __restrict__ outp,
                                              int rowBase, int N,
                                              int lsel, int lrow)
{
    const int rowOff = lsel * 8;
    const int k0a = (lsel ? 8 : 0);
    const int k1a = (lsel ? 24 : 16);
    for (int t = 0; t < TM / 16; ++t) {
        const int R = t * 16;
        // A fragments (16-bit A 16x32 layout): two b128 LDS loads per kblock
        v16h afr[2];
#pragma unroll
        for (int kb = 0; kb < 2; ++kb) {
            const v8h lo = *(const v8h*)&sX[(R + lrow) * DIM + 32 * kb + k0a];
            const v8h hi = *(const v8h*)&sX[(R + lrow) * DIM + 32 * kb + k1a];
#pragma unroll
            for (int i = 0; i < 8; ++i) { afr[kb][i] = lo[i]; afr[kb][i + 8] = hi[i]; }
        }
#pragma unroll
        for (int nt = 0; nt < 4; ++nt) {
            v8f acc = {};
            acc = __builtin_amdgcn_wmma_f32_16x16x32_f16(false, afr[0], false, bfr[nt][0],
                                                         (short)0, acc, false, false);
            acc = __builtin_amdgcn_wmma_f32_16x16x32_f16(false, afr[1], false, bfr[nt][1],
                                                         (short)0, acc, false, false);
            // C/D layout: lane<16 -> (M=r, N=lane); lane>=16 -> (M=8+r, N=lane-16)
            const int gcol = nt * 16 + lrow;
            const float bb = bv4[nt];
#pragma unroll
            for (int r = 0; r < 8; ++r) {
                int grow = rowBase + R + r + rowOff;
                if (FULL || grow < N)
                    outp[(size_t)grow * DIM + gcol] = acc[r] + bb;
            }
        }
    }
}

// ---------------------------------------------------------------------------
// Fused projection GEMM: q = x@Wq+bq, k = x@Wk+bk, v = x@Wv+bv, out = x@Ws+bs.
// Block = 4 waves; wave w owns matrix w. x tile staged once in LDS (f16,
// shared by all 4 waves); B fragments loaded coalesced (b128) straight from
// the pre-transposed f16 weights in L2. ReLU of the previous layer is fused
// into the x staging (relu_in).
// ---------------------------------------------------------------------------
__global__ __launch_bounds__(128)
void gemm_qkvs(const float* __restrict__ x, int N, int relu_in,
               const _Float16* __restrict__ Wt16,
               const float* __restrict__ b0, const float* __restrict__ b1,
               const float* __restrict__ b2, const float* __restrict__ b3,
               float* __restrict__ o0, float* __restrict__ o1,
               float* __restrict__ o2, float* __restrict__ o3)
{
    __shared__ __align__(16) _Float16 sX[TM * DIM];   // 16 KB

    const int tid  = threadIdx.x;
    const int wave = tid >> 5;
    const int lane = tid & 31;
    const int rowBase = blockIdx.x * TM;
    const bool fullBlk = (rowBase + TM <= N);

    // stage x tile (f32 -> f16), float4 coalesced, optional fused ReLU
    for (int i = tid * 4; i < TM * DIM; i += 128 * 4) {
        int r = i >> 6, c = i & 63;
        int gr = rowBase + r;
        float4 val = make_float4(0.f, 0.f, 0.f, 0.f);
        if (gr < N) val = *(const float4*)&x[(size_t)gr * DIM + c];
        if (relu_in) {
            val.x = fmaxf(val.x, 0.f); val.y = fmaxf(val.y, 0.f);
            val.z = fmaxf(val.z, 0.f); val.w = fmaxf(val.w, 0.f);
        }
        sX[i + 0] = (_Float16)val.x; sX[i + 1] = (_Float16)val.y;
        sX[i + 2] = (_Float16)val.z; sX[i + 3] = (_Float16)val.w;
    }
    __syncthreads();

    float* outp = (wave == 0) ? o0 : (wave == 1) ? o1 : (wave == 2) ? o2 : o3;
    const float* bia = (wave == 0) ? b0 : (wave == 1) ? b1 : (wave == 2) ? b2 : b3;
    const _Float16* Wt = Wt16 + wave * (DIM * DIM);

    const int lsel = lane >> 4;        // 0: lanes 0-15, 1: lanes 16-31
    const int lrow = lane & 15;

    // B fragments: invariant across row tiles -> load once (global b128, L2-hot).
    // 16-bit B 32x16 layout: lane<16 col=lane, K halves {0..7},{16..23};
    // lane>=16 col=lane-16, K halves {8..15},{24..31} (+32*kblock).
    v16h  bfr[4][2];
    float bv4[4];
#pragma unroll
    for (int nt = 0; nt < 4; ++nt) {
        bv4[nt] = bia[nt * 16 + lrow];
#pragma unroll
        for (int kb = 0; kb < 2; ++kb) {
            int kbase = 32 * kb;
            int k0 = kbase + (lsel ? 8 : 0);
            int k1 = kbase + (lsel ? 24 : 16);
            const v8h lo = *(const v8h*)&Wt[(nt * 16 + lrow) * DIM + k0];
            const v8h hi = *(const v8h*)&Wt[(nt * 16 + lrow) * DIM + k1];
#pragma unroll
            for (int i = 0; i < 8; ++i) { bfr[nt][kb][i] = lo[i]; bfr[nt][kb][i + 8] = hi[i]; }
        }
    }

    if (fullBlk) compute_tiles<true >(sX, bfr, bv4, outp, rowBase, N, lsel, lrow);
    else         compute_tiles<false>(sX, bfr, bv4, outp, rowBase, N, lsel, lrow);
}

// ---------------------------------------------------------------------------
__device__ inline void atomicMaxFloat(float* addr, float val) {
    if (val >= 0.0f) atomicMax((int*)addr, __float_as_int(val));
    else             atomicMin((unsigned int*)addr, (unsigned int)__float_as_int(val));
}

__global__ void init_nodes(float* __restrict__ mmax, float* __restrict__ denom, int N) {
    int i = blockIdx.x * blockDim.x + threadIdx.x;
    if (i < N) { mmax[i] = -__builtin_inff(); denom[i] = 0.0f; }
}

// 16 lanes per edge: float4 gathers of q[dst], k[src]; wave32 shfl reduction.
__global__ void edge_logits(const float* __restrict__ q, const float* __restrict__ k,
                            const int* __restrict__ src, const int* __restrict__ dst,
                            int E, float* __restrict__ logit, float* __restrict__ mmax)
{
    int t = blockIdx.x * blockDim.x + threadIdx.x;
    int e = t >> 4;
    int sub = t & 15;
    bool valid = (e < E);
    int ec = valid ? e : (E - 1);
    int s = src[ec], d = dst[ec];
    const float4 qv = *(const float4*)&q[(size_t)d * DIM + sub * 4];
    const float4 kv = *(const float4*)&k[(size_t)s * DIM + sub * 4];
    float p = qv.x * kv.x + qv.y * kv.y + qv.z * kv.z + qv.w * kv.w;
    p += __shfl_xor(p, 8);
    p += __shfl_xor(p, 4);
    p += __shfl_xor(p, 2);
    p += __shfl_xor(p, 1);
    if (valid && sub == 0) {
        p *= 0.125f;   // 1/sqrt(64)
        logit[e] = p;
        atomicMaxFloat(&mmax[d], p);
    }
}

__global__ void edge_exp(const int* __restrict__ dst, int E,
                         float* __restrict__ logit, const float* __restrict__ mmax,
                         float* __restrict__ denom)
{
    int e = blockIdx.x * blockDim.x + threadIdx.x;
    if (e >= E) return;
    int d = dst[e];
    float ee = __expf(logit[e] - mmax[d]);
    logit[e] = ee;             // reuse buffer for exp values
    atomicAdd(&denom[d], ee);
}

// 16 lanes per edge: atomicAdd alpha * v[src] into out[dst]
__global__ void edge_scatter(const int* __restrict__ src, const int* __restrict__ dst,
                             int E, const float* __restrict__ ebuf,
                             const float* __restrict__ denom,
                             const float* __restrict__ v, float* __restrict__ outx)
{
    int t = blockIdx.x * blockDim.x + threadIdx.x;
    int e = t >> 4;
    if (e >= E) return;
    int sub = t & 15;
    int s = src[e], d = dst[e];
    float alpha = ebuf[e] / denom[d];
    const float4 vv = *(const float4*)&v[(size_t)s * DIM + sub * 4];
    float* op = &outx[(size_t)d * DIM + sub * 4];
    atomicAdd(op + 0, alpha * vv.x);
    atomicAdd(op + 1, alpha * vv.y);
    atomicAdd(op + 2, alpha * vv.z);
    atomicAdd(op + 3, alpha * vv.w);
}

__global__ void relu_k(float* __restrict__ x, int n) {
    int i = blockIdx.x * blockDim.x + threadIdx.x;
    if (i < n) x[i] = fmaxf(x[i], 0.0f);
}

__global__ void pool_zero(float* __restrict__ gsum, float* __restrict__ gcnt, int G) {
    int i = blockIdx.x * blockDim.x + threadIdx.x;
    if (i < G * DIM) gsum[i] = 0.0f;
    if (i < G) gcnt[i] = 0.0f;
}

__global__ void pool_accum(const float* __restrict__ x, const int* __restrict__ batch,
                           int N, float* __restrict__ gsum, float* __restrict__ gcnt)
{
    int t = blockIdx.x * blockDim.x + threadIdx.x;
    int n = t >> 6, j = t & 63;
    if (n >= N) return;
    int g = batch[n];
    atomicAdd(&gsum[(size_t)g * DIM + j], x[(size_t)n * DIM + j]);
    if (j == 0) atomicAdd(&gcnt[g], 1.0f);
}

__global__ void pool_div(const float* __restrict__ gsum, const float* __restrict__ gcnt,
                         int G, float* __restrict__ out)
{
    int i = blockIdx.x * blockDim.x + threadIdx.x;
    if (i < G * DIM) out[i] = gsum[i] / fmaxf(gcnt[i >> 6], 1.0f);
}

// ---------------------------------------------------------------------------
extern "C" void kernel_launch(void* const* d_in, const int* in_sizes, int n_in,
                              void* d_out, int out_size, void* d_ws, size_t ws_size,
                              hipStream_t stream)
{
    const float* x0    = (const float*)d_in[0];
    const int*   ei    = (const int*)d_in[1];
    const int*   batch = (const int*)d_in[2];
    const float* Wq = (const float*)d_in[3];
    const float* bq = (const float*)d_in[4];
    const float* Wk = (const float*)d_in[5];
    const float* bk = (const float*)d_in[6];
    const float* Wv = (const float*)d_in[7];
    const float* bv = (const float*)d_in[8];
    const float* Ws = (const float*)d_in[9];
    const float* bs = (const float*)d_in[10];

    const int N = in_sizes[0] / DIM;
    const int E = in_sizes[1] / 2;
    const int G = out_size / DIM - N;
    const int L = 3;

    const int* src = ei;
    const int* dst = ei + E;

    // workspace layout (floats)
    float* ws = (float*)d_ws;
    size_t nd = (size_t)N * DIM;
    float* xA    = ws;
    float* xB    = xA + nd;
    float* qb    = xB + nd;
    float* kb    = qb + nd;
    float* vb    = kb + nd;
    float* logit = vb + nd;                 // E floats (logits, then exp values)
    float* mmax  = logit + E;               // N
    float* denom = mmax + N;                // N
    float* gsum  = denom + N;               // G*DIM
    float* gcnt  = gsum + (size_t)G * DIM;  // G
    // f16 transposed weights for current layer (16B aligned)
    size_t wtOff = (size_t)(gcnt + G - ws);
    wtOff = (wtOff + 7) & ~(size_t)7;
    _Float16* wt16 = (_Float16*)(ws + wtOff);   // 4*64*64 halves = 32 KB

    float* x_final = (float*)d_out;              // [N, D]
    float* g_out   = x_final + nd;               // [G, D]

    const float* xin = x0;
    for (int l = 0; l < L; ++l) {
        float* xout = (l == 0) ? xA : (l == 1) ? xB : x_final;

        prep_weights<<<(4 * DIM * DIM + 255) / 256, 256, 0, stream>>>(
            Wq + (size_t)l * DIM * DIM, Wk + (size_t)l * DIM * DIM,
            Wv + (size_t)l * DIM * DIM, Ws + (size_t)l * DIM * DIM, wt16);

        gemm_qkvs<<<(N + TM - 1) / TM, 128, 0, stream>>>(
            xin, N, (l > 0) ? 1 : 0, wt16,
            bq + (size_t)l * DIM, bk + (size_t)l * DIM,
            bv + (size_t)l * DIM, bs + (size_t)l * DIM,
            qb, kb, vb, xout);

        init_nodes<<<(N + 255) / 256, 256, 0, stream>>>(mmax, denom, N);

        int lanesE = E * 16;
        edge_logits<<<(lanesE + 255) / 256, 256, 0, stream>>>(qb, kb, src, dst, E, logit, mmax);
        edge_exp<<<(E + 255) / 256, 256, 0, stream>>>(dst, E, logit, mmax, denom);
        edge_scatter<<<(lanesE + 255) / 256, 256, 0, stream>>>(src, dst, E, logit, denom, vb, xout);

        xin = xout;
    }

    // final ReLU on d_out x-region, then mean pool
    relu_k<<<((int)nd + 255) / 256, 256, 0, stream>>>(x_final, (int)nd);
    pool_zero<<<(G * DIM + 255) / 256, 256, 0, stream>>>(gsum, gcnt, G);
    pool_accum<<<((int)nd + 255) / 256, 256, 0, stream>>>(x_final, batch, N, gsum, gcnt);
    pool_div<<<(G * DIM + 255) / 256, 256, 0, stream>>>(gsum, gcnt, G, g_out);
}
